// TimelineGNNLayer3_39410619908398
// MI455X (gfx1250) — matrix operand order repaired and checked by reading, not compiled, not run on hardware
//
#include <hip/hip_runtime.h>
#include <math.h>

// ---------------------------------------------------------------------------
// Types for CDNA5 WMMA (wave32): D(16x16 f32) = A(16x32 bf16) x B(32x16 bf16) + C
// ---------------------------------------------------------------------------
typedef __attribute__((ext_vector_type(16))) __bf16 v16bf;
typedef __attribute__((ext_vector_type(8)))  float  v8f;

#define N_NODE   50000
#define N_EDGE   500000
#define DIM      128
#define EDGE_TILE 16

__device__ __forceinline__ unsigned short f2bf(float f) {
    unsigned int u = __float_as_uint(f);
    return (unsigned short)((u + 0x7FFFu + ((u >> 16) & 1u)) >> 16);  // RNE
}
__device__ __forceinline__ float bf2f(unsigned short h) {
    return __uint_as_float(((unsigned int)h) << 16);
}
__device__ __forceinline__ float lrelu(float x) { return x > 0.0f ? x : 0.01f * x; }
__device__ __forceinline__ float sigmoidf(float x) { return 1.0f / (1.0f + __expf(-x)); }

// ---------------------------------------------------------------------------
// One wave computes a 16(M) x 16(N) f32 tile, accumulating over nKt K-chunks
// of 32. A lives in LDS as bf16 rows (stride strideA elements). B fragments
// are pre-swizzled in global memory: frag (kt,nt) = 32 lanes x 16 bf16,
// contiguous per lane.
// A layout (ISA 7.12.2, 16-bit A 16x32): lane L -> row m=L&15; lanes>=16 get
// K-base +8; vector elems 0..7 = K kb..kb+7, elems 8..15 = K kb+16..kb+23.
// ---------------------------------------------------------------------------
__device__ __forceinline__ v8f gemm_acc(v8f c,
                                        const unsigned short* __restrict__ A, int strideA,
                                        const unsigned short* __restrict__ F, int nN, int nt,
                                        int nKt, int lane) {
    const int m    = lane & 15;
    const int koff = (lane & 16) ? 8 : 0;
    for (int kt = 0; kt < nKt; ++kt) {
        v16bf a, b;
        const unsigned short* ap = A + m * strideA + kt * 32 + koff;
        ((uint4*)&a)[0] = *(const uint4*)(ap);
        ((uint4*)&a)[1] = *(const uint4*)(ap + 16);
        const unsigned short* bp = F + (((size_t)(kt * nN + nt)) << 9) + lane * 16;
        ((uint4*)&b)[0] = *(const uint4*)(bp);
        ((uint4*)&b)[1] = *(const uint4*)(bp + 8);
        c = __builtin_amdgcn_wmma_f32_16x16x32_bf16(false, a, false, b, (short)0, c,
                                                    false, false);
    }
    return c;
}

// ---------------------------------------------------------------------------
// Weight prep: W is (Nout x K) f32 row-major (the reference computes x @ W.T,
// so B[k][n] = W[n][k]). Emit bf16 B-fragments, ktile-major:
//   frag index = ktile*nN + ntile; within frag: lane (0..31) x elem (0..15);
//   n = ntile*16 + (lane&15); k = ktile*32 + (lane&16 ? 8 : 0) + (elem<8 ? elem : elem+8).
// ---------------------------------------------------------------------------
__global__ void prep_frags(const float* __restrict__ W, unsigned short* __restrict__ dst,
                           int K, int nN, int nK) {
    int idx = blockIdx.x * 256 + threadIdx.x;
    int total = nN * nK * 512;
    if (idx >= total) return;
    int f = idx >> 9, r = idx & 511;
    int lane = r >> 4, e = r & 15;
    int ktile = f / nN, ntile = f % nN;
    int n = ntile * 16 + (lane & 15);
    int k = ktile * 32 + ((lane & 16) ? 8 : 0) + (e < 8 ? e : e + 8);
    dst[idx] = f2bf(W[(size_t)n * K + k]);
}

__global__ void zero_f32(float* __restrict__ p, size_t n) {
    size_t i = (size_t)blockIdx.x * blockDim.x + threadIdx.x;
    size_t stride = (size_t)gridDim.x * blockDim.x;
    for (; i < n; i += stride) p[i] = 0.0f;
}

// ---------------------------------------------------------------------------
// Main per-edge kernel. 256 threads = 8 waves; each block handles 16 edges.
// LDS activation buffer s_in rows = [hr(0..127) | h_qr(128..255) | hs(256..383)]
// (cols 0..127 start as rel_emb, overwritten by hr after fuse2).
// ---------------------------------------------------------------------------
__global__ __launch_bounds__(256)
void edge_kernel(const int* __restrict__ edges, const int* __restrict__ q_rel,
                 const float* __restrict__ hidden, const float* __restrict__ rela_w,
                 const float* __restrict__ fuse1_b, const float* __restrict__ fuse2_b,
                 const float* __restrict__ gate_b, const float* __restrict__ htrans_b,
                 const float* __restrict__ wqr_b, const float* __restrict__ walpha,
                 const unsigned short* __restrict__ Ff1, const unsigned short* __restrict__ Ff2,
                 const unsigned short* __restrict__ Fg,  const unsigned short* __restrict__ Fht,
                 const unsigned short* __restrict__ Fatt,
                 float* __restrict__ up, float* __restrict__ bottom) {
    __shared__ unsigned short s_in[EDGE_TILE * 384];   // bf16 [hr|hqr|hs]
    __shared__ unsigned short s_time[EDGE_TILE * 32];  // bf16 time emb
    __shared__ unsigned short s_x[EDGE_TILE * 128];    // bf16 fuse1 out
    __shared__ unsigned short s_rs[EDGE_TILE * 128];   // bf16 reset*hs
    __shared__ float s_upd[EDGE_TILE * 128];           // update gate
    __shared__ float s_msg[EDGE_TILE * 128];           // message
    __shared__ float s_alpha[EDGE_TILE];
    __shared__ float s_ea[EDGE_TILE];
    __shared__ int   s_obj[EDGE_TILE];

    const int tid  = threadIdx.x;
    const int lane = tid & 31;
    const int w    = tid >> 5;          // wave id 0..7 -> N-tile
    const int m16  = tid >> 4;          // edge row 0..15 for per-row work
    const int l16  = tid & 15;
    const int ebase = blockIdx.x * EDGE_TILE;

    // ---- gather edge data into LDS (bf16) ----
    {
        const int* er = edges + (size_t)(ebase + m16) * 7;
        const int rel = er[2];
        const int sub = er[4];
        const int rq  = er[0];
        const int tt  = er[6];
        const int hqr_row = q_rel[rq];
        #pragma unroll
        for (int j = 0; j < 8; ++j) {
            int col = l16 * 8 + j;
            s_in[m16 * 384 + col]       = f2bf(rela_w[(size_t)rel * DIM + col]);
            s_in[m16 * 384 + 128 + col] = f2bf(rela_w[(size_t)hqr_row * DIM + col]);
            s_in[m16 * 384 + 256 + col] = f2bf(hidden[(size_t)sub * DIM + col]);
        }
        if (l16 < 4) {
            #pragma unroll
            for (int j = 0; j < 8; ++j) {
                int cc = l16 * 8 + j;                       // 0..31
                float d   = __expf(-(float)(cc & ~1) * (9.210340371976184f / 32.0f));
                float arg = (float)tt * d;
                float v   = (cc & 1) ? __cosf(arg) : __sinf(arg);
                s_time[m16 * 32 + cc] = f2bf(v);
            }
        }
        if (tid < EDGE_TILE) {
            s_obj[tid]   = edges[(size_t)(ebase + tid) * 7 + 5];
            s_alpha[tid] = 0.0f;
        }
    }
    __syncthreads();

    const int ncol = w * 16 + (lane & 15);
    const int mhi  = (lane & 16) ? 8 : 0;
    v8f zero8 = {};

    // ---- fuse1: lrelu([rel|time] @ fuse1_w.T + b)  (K=160: 4 chunks rel + 1 time) ----
    {
        v8f c = gemm_acc(zero8, s_in, 384, Ff1, 8, w, 4, lane);
        c = gemm_acc(c, s_time, 32, Ff1 + 4 * 8 * 512, 8, w, 1, lane);
        float bsv = fuse1_b[ncol];
        #pragma unroll
        for (int v = 0; v < 8; ++v) {
            int mm = v + mhi;
            s_x[mm * 128 + ncol] = f2bf(lrelu(c[v] + bsv));
        }
    }
    __syncthreads();

    // ---- fuse2 -> hr = lrelu(.)+rel_emb; store hr over cols 0..127 of s_in ----
    {
        v8f c = gemm_acc(zero8, s_x, 128, Ff2, 8, w, 4, lane);
        float bsv = fuse2_b[ncol];
        #pragma unroll
        for (int v = 0; v < 8; ++v) {
            int mm = v + mhi;
            float hr = lrelu(c[v] + bsv) + bf2f(s_in[mm * 384 + ncol]);
            s_in[mm * 384 + ncol] = f2bf(hr);
        }
    }
    __syncthreads();

    // ---- gate: sigmoid([hr|hqr|hs] @ gate_w.T + b), N=256 -> 2 tiles per wave ----
    #pragma unroll
    for (int half = 0; half < 2; ++half) {
        int nt = w + half * 8;
        v8f c = gemm_acc(zero8, s_in, 384, Fg, 16, nt, 12, lane);
        int gcol = nt * 16 + (lane & 15);
        float bsv = gate_b[gcol];
        #pragma unroll
        for (int v = 0; v < 8; ++v) {
            int mm = v + mhi;
            float g = sigmoidf(c[v] + bsv);
            if (gcol < 128) {
                s_upd[mm * 128 + gcol] = g;                          // update
            } else {
                int cr = gcol - 128;                                 // reset
                float hsv = bf2f(s_in[mm * 384 + 256 + cr]);
                s_rs[mm * 128 + cr] = f2bf(g * hsv);
            }
        }
    }
    __syncthreads();

    // ---- attention pre-act: [hr|hqr|hs] @ [wr;wqr;ws].T + wqr_b, folded walpha dot ----
    {
        v8f c = gemm_acc(zero8, s_in, 384, Fatt, 8, w, 12, lane);
        float bsv = wqr_b[ncol];
        float wa  = walpha[ncol];
        #pragma unroll
        for (int v = 0; v < 8; ++v) {
            int mm = v + mhi;
            atomicAdd(&s_alpha[mm], lrelu(c[v] + bsv) * wa);
        }
    }
    __syncthreads();

    // ---- move reset*hs into s_in cols 128..255 for htrans ----
    #pragma unroll
    for (int j = 0; j < 8; ++j) {
        int col = l16 * 8 + j;
        s_in[m16 * 384 + 128 + col] = s_rs[m16 * 128 + col];
    }
    __syncthreads();

    // ---- htrans: tanh([hr|reset*hs] @ htrans_w.T + b); message = (1-u)*hs + u*cand ----
    {
        v8f c = gemm_acc(zero8, s_in, 384, Fht, 8, w, 8, lane);
        float bsv = htrans_b[ncol];
        #pragma unroll
        for (int v = 0; v < 8; ++v) {
            int mm = v + mhi;
            float cand = tanhf(c[v] + bsv);
            float u    = s_upd[mm * 128 + ncol];
            float hsv  = bf2f(s_in[mm * 384 + 256 + ncol]);
            s_msg[mm * 128 + ncol] = (1.0f - u) * hsv + u * cand;
        }
    }
    __syncthreads();

    if (tid < EDGE_TILE) s_ea[tid] = __expf(s_alpha[tid]);
    __syncthreads();

    // ---- segment-sum scatter: up[obj] += e*msg ; bottom[obj] += e ----
    {
        int   obj = s_obj[m16];
        float e   = s_ea[m16];
        #pragma unroll
        for (int j = 0; j < 8; ++j) {
            int col = l16 * 8 + j;
            atomicAdd(&up[(size_t)obj * DIM + col], e * s_msg[m16 * 128 + col]);
        }
        if (l16 == 0) atomicAdd(&bottom[obj], e);
    }
}

// ---------------------------------------------------------------------------
// Final: out = (up / (bottom+1e-5)) @ wh_w.T over 16-node tiles.
// ---------------------------------------------------------------------------
__global__ __launch_bounds__(256)
void agg_out_kernel(const float* __restrict__ up, const float* __restrict__ bottom,
                    const unsigned short* __restrict__ Fwh, float* __restrict__ out) {
    __shared__ unsigned short s_a[16 * 128];
    const int tid = threadIdx.x;
    const int m16 = tid >> 4, l16 = tid & 15;
    const int nbase = blockIdx.x * 16;

    float bsum = bottom[nbase + m16] + 1e-5f;
    float inv = 1.0f / bsum;
    #pragma unroll
    for (int j = 0; j < 8; ++j) {
        int col = l16 * 8 + j;
        s_a[m16 * 128 + col] = f2bf(up[(size_t)(nbase + m16) * DIM + col] * inv);
    }
    __syncthreads();

    const int lane = tid & 31, w = tid >> 5;
    v8f zero8 = {};
    v8f c = gemm_acc(zero8, s_a, 128, Fwh, 8, w, 4, lane);
    const int ncol = w * 16 + (lane & 15);
    const int mhi  = (lane & 16) ? 8 : 0;
    #pragma unroll
    for (int v = 0; v < 8; ++v) {
        int mm = v + mhi;
        out[(size_t)(nbase + mm) * DIM + ncol] = c[v];
    }
}

// ---------------------------------------------------------------------------
extern "C" void kernel_launch(void* const* d_in, const int* in_sizes, int n_in,
                              void* d_out, int out_size, void* d_ws, size_t ws_size,
                              hipStream_t stream) {
    // input order: q_sub(0) q_rel(1) hidden(2) edges(3) n_node(4) rela_w(5)
    // fuse1_w(6) fuse1_b(7) fuse2_w(8) fuse2_b(9) gate_w(10) gate_b(11)
    // htrans_w(12) htrans_b(13) ws_w(14) wr_w(15) wqr_w(16) wqr_b(17)
    // walpha_w(18) wh_w(19)
    const int*   q_rel    = (const int*)d_in[1];
    const float* hidden   = (const float*)d_in[2];
    const int*   edges    = (const int*)d_in[3];
    const float* rela_w   = (const float*)d_in[5];
    const float* fuse1_w  = (const float*)d_in[6];
    const float* fuse1_b  = (const float*)d_in[7];
    const float* fuse2_w  = (const float*)d_in[8];
    const float* fuse2_b  = (const float*)d_in[9];
    const float* gate_w   = (const float*)d_in[10];
    const float* gate_b   = (const float*)d_in[11];
    const float* htrans_w = (const float*)d_in[12];
    const float* htrans_b = (const float*)d_in[13];
    const float* ws_w     = (const float*)d_in[14];
    const float* wr_w     = (const float*)d_in[15];
    const float* wqr_w    = (const float*)d_in[16];
    const float* wqr_b    = (const float*)d_in[17];
    const float* walpha   = (const float*)d_in[18];
    const float* wh_w     = (const float*)d_in[19];
    float* out = (float*)d_out;

    // workspace layout
    float* up     = (float*)d_ws;                     // N_NODE*128
    float* bottom = up + (size_t)N_NODE * DIM;        // N_NODE (contiguous after up)
    size_t off = (((size_t)N_NODE * DIM + N_NODE) * sizeof(float) + 255) & ~(size_t)255;
    unsigned short* F   = (unsigned short*)((char*)d_ws + off);
    unsigned short* Ff1 = F;  F += (size_t)5  * 8  * 512;   // fuse1: K=160
    unsigned short* Ff2 = F;  F += (size_t)4  * 8  * 512;   // fuse2: K=128
    unsigned short* Fg  = F;  F += (size_t)12 * 16 * 512;   // gate:  K=384,N=256
    unsigned short* Fht = F;  F += (size_t)8  * 8  * 512;   // htrans:K=256
    unsigned short* Fatt= F;  F += (size_t)12 * 8  * 512;   // [wr|wqr|ws]: K=384
    unsigned short* Fwh = F;  F += (size_t)4  * 8  * 512;   // wh: K=128

    // weight fragment prep (bf16, WMMA B layout)
    auto blocks = [](int total) { return (total + 255) / 256; };
    prep_frags<<<blocks(5  * 8  * 512), 256, 0, stream>>>(fuse1_w,  Ff1, 160, 8, 5);
    prep_frags<<<blocks(4  * 8  * 512), 256, 0, stream>>>(fuse2_w,  Ff2, 128, 8, 4);
    prep_frags<<<blocks(12 * 16 * 512), 256, 0, stream>>>(gate_w,   Fg,  384, 16, 12);
    prep_frags<<<blocks(8  * 8  * 512), 256, 0, stream>>>(htrans_w, Fht, 256, 8, 8);
    prep_frags<<<blocks(4  * 8  * 512), 256, 0, stream>>>(wr_w,  Fatt,            128, 8, 4);
    prep_frags<<<blocks(4  * 8  * 512), 256, 0, stream>>>(wqr_w, Fatt + 32 * 512, 128, 8, 4);
    prep_frags<<<blocks(4  * 8  * 512), 256, 0, stream>>>(ws_w,  Fatt + 64 * 512, 128, 8, 4);
    prep_frags<<<blocks(4  * 8  * 512), 256, 0, stream>>>(wh_w,  Fwh,             128, 8, 4);

    // zero segment-sum accumulators (up + bottom are contiguous)
    zero_f32<<<2048, 256, 0, stream>>>(up, (size_t)N_NODE * DIM + N_NODE);

    // per-edge pipeline: 500000 / 16 = 31250 blocks
    edge_kernel<<<N_EDGE / EDGE_TILE, 256, 0, stream>>>(
        edges, q_rel, hidden, rela_w, fuse1_b, fuse2_b, gate_b, htrans_b,
        wqr_b, walpha, Ff1, Ff2, Fg, Fht, Fatt, up, bottom);

    // aggregation + output GEMM: 50000 / 16 = 3125 blocks
    agg_out_kernel<<<N_NODE / 16, 256, 0, stream>>>(up, bottom, Fwh, out);
}
